// LinearSelfAttention_29721173688716
// MI455X (gfx1250) — compile-verified, matrix operand order checked by reading, not run on hardware
//
#include <hip/hip_runtime.h>

// ---------------------------------------------------------------------------
// Linformer self-attention for MI455X (gfx1250), wave32 + WMMA f32_16x16x32_f16
// + GLOBAL_LOAD_ASYNC_TO_LDS_B128 for f16 operand tiles, double-buffered LDS.
//
// Pipeline (all GEMMs share one WMMA kernel in "out = A @ Bt^T" form, both
// operands contiguous along K):
//   1. Et = E^T, Ft = F^T                    (f32 -> f16 transpose)
//   2. Qh = q@W_Q^T            [B,S,D] f16   (normal store)
//      Kt = (k@W_K^T)^T        [B,D,S] f16   (transposed store)
//      Vt = (v@W_V^T)^T        [B,D,S] f16   (transposed store)
//   3. Kpt[z] = Et @ Kt[z]^T   [256,64] f16   per z=(b,h)   (== Kp^T)
//      Vpt[z] = Vt[z] @ Ft^T   [64,256] f16   per z         (== Vp^T)
//   4. Sc[z]  = (Qh[z] @ Kpt[z]^T) * 1/8  [4096,256] f16
//   5. softmax rows of 256 in place (f32 math, wave32 shfl reductions)
//   6. attn   = Sc @ Vpt^T -> attnAll [B,S,D] f16 (head-interleaved)
//   7. out    = attnAll @ W_O^T + b_O  -> f32
//
// Out-of-range tile rows are handled by CLAMPING the row index (no EXEC
// divergence, no zero-fill): clamped rows only feed C fragments whose stores
// are guarded, so they never reach the output.
// ---------------------------------------------------------------------------

typedef _Float16 half_t;
typedef _Float16 v16h __attribute__((ext_vector_type(16)));
typedef _Float16 v8h  __attribute__((ext_vector_type(8)));
typedef _Float16 v4h  __attribute__((ext_vector_type(4)));
typedef float    v8f  __attribute__((ext_vector_type(8)));
typedef float    v4f  __attribute__((ext_vector_type(4)));

#define BM 128
#define BN 128
#define BK 32
#define LDSS (BK + 8)   // 40 halves = 80B row stride: 16B aligned, bank-conflict-free

// ---- staging descriptors ---------------------------------------------------
struct StageF32   { v4f v[4]; };   // f32 path: global -> VGPR -> cvt -> LDS
struct StageAsync { };             // f16 path: global -> LDS async DMA

// f32 operand: issue all 4 unconditional v4f loads (row-clamped)
__device__ __forceinline__ void stage(const float* __restrict__ g, int rows, int ld,
                                      int rowBase, int kBase, half_t* /*lds*/,
                                      int tid, StageF32& st)
{
#pragma unroll
    for (int i = 0; i < 4; ++i) {            // 128 rows * 8 float4-chunks / 256 thr
        int chunk = tid + (i << 8);
        int r  = chunk >> 3;
        int c4 = (chunk & 7) << 2;
        int grow = rowBase + r;
        grow = grow < rows ? grow : rows - 1;   // clamp, always in-bounds
        st.v[i] = *(const v4f*)(g + (long long)grow * ld + kBase + c4);
    }
}
__device__ __forceinline__ void commit(half_t* lds, int tid, StageF32& st)
{
#pragma unroll
    for (int i = 0; i < 4; ++i) {
        int chunk = tid + (i << 8);
        int r  = chunk >> 3;
        int c4 = (chunk & 7) << 2;
        v4h h;
        h[0] = (half_t)st.v[i][0]; h[1] = (half_t)st.v[i][1];
        h[2] = (half_t)st.v[i][2]; h[3] = (half_t)st.v[i][3];
        *(v4h*)(lds + r * LDSS + c4) = h;       // 8B aligned
    }
}

// f16 operand: per-lane async DMA straight into LDS (16B per lane),
// tracked with ASYNCcnt.  All 256 threads active -> EXEC all ones.
__device__ __forceinline__ void stage(const half_t* __restrict__ g, int rows, int ld,
                                      int rowBase, int kBase, half_t* lds,
                                      int tid, StageAsync&)
{
#pragma unroll
    for (int i = 0; i < 2; ++i) {            // 128 rows * 4 v8h-chunks / 256 thr
        int chunk = tid + (i << 8);
        int r  = chunk >> 2;
        int c8 = (chunk & 3) << 3;
        int grow = rowBase + r;
        grow = grow < rows ? grow : rows - 1;   // clamp, always in-bounds
        unsigned long long gaddr =
            (unsigned long long)(size_t)(g + (long long)grow * ld + kBase + c8);
        unsigned ldsaddr = (unsigned)(size_t)(lds + r * LDSS + c8); // flat[31:0] = LDS offset
        asm volatile("global_load_async_to_lds_b128 %0, %1, off"
                     :: "v"(ldsaddr), "v"(gaddr)
                     : "memory");
    }
}
__device__ __forceinline__ void commit(half_t*, int, StageAsync&) {}

template <typename T> struct StageFor;
template <> struct StageFor<float>  { using type = StageF32; };
template <> struct StageFor<half_t> { using type = StageAsync; };

// ---- LDS -> WMMA operand fragment (ISA 16-bit A/B layout) ------------------
// lane<16 : row = base+lane,    halves = [k0..7 | k16..23]
// lane>=16: row = base+lane-16, halves = [k8..15 | k24..31]
__device__ __forceinline__ v16h load_frag(const half_t* lds, int rowBase, int lane)
{
    int r  = rowBase + (lane & 15);
    int ko = (lane >> 4) << 3;
    const half_t* p = lds + r * LDSS + ko;
    v8h lo = *(const v8h*)(p);
    v8h hi = *(const v8h*)(p + 16);
    v16h a;
#pragma unroll
    for (int i = 0; i < 8; ++i) { a[i] = lo[i]; a[i + 8] = hi[i]; }
    return a;
}

// ---- generic batched WMMA GEMM:  out = (A @ Bt^T + bias) * outScale --------
// A: [M,K] lda, row-contiguous-K.  Bt: [N,K] ldb, row-contiguous-K.
// OUTMODE: 0 = f32 normal, 1 = f16 normal, 2 = f16 transposed (O[n*ldo+m]).
// Batch z = blockIdx.z: outer = z / batchInner, inner = z % batchInner.
template <typename TA, typename TB, int OUTMODE>
__launch_bounds__(256)
__global__ void gemm_wmma(const TA* __restrict__ Ag, const TB* __restrict__ Bg,
                          void* __restrict__ Og, const float* __restrict__ bias,
                          int M, int N, int K, int lda, int ldb, int ldo,
                          int batchInner,
                          long long sA0, long long sA1,
                          long long sB0, long long sB1,
                          long long sO0, long long sO1,
                          float outScale)
{
    __shared__ half_t As[2][BM * LDSS];
    __shared__ half_t Bs[2][BN * LDSS];

    const int tid  = threadIdx.x;
    const int lane = tid & 31;
    const int w    = tid >> 5;           // 8 waves
    const int wm   = (w & 1) * 64;       // wave tile: 64 (M) x 32 (N)
    const int wn   = (w >> 1) * 32;
    const int bM   = blockIdx.y * BM;
    const int bN   = blockIdx.x * BN;

    const int z     = blockIdx.z;
    const int outer = z / batchInner;
    const int inner = z - outer * batchInner;
    const TA* Ab = Ag + (long long)outer * sA0 + (long long)inner * sA1;
    const TB* Bb = Bg + (long long)outer * sB0 + (long long)inner * sB1;
    const long long oOff = (long long)outer * sO0 + (long long)inner * sO1;

    typename StageFor<TA>::type ra;
    typename StageFor<TB>::type rb;

    v8f c[4][2];
#pragma unroll
    for (int i = 0; i < 4; ++i)
#pragma unroll
        for (int j = 0; j < 2; ++j)
#pragma unroll
            for (int e = 0; e < 8; ++e) c[i][j][e] = 0.0f;

    // prologue: tile 0 into buffer 0
    stage(Ab, M, lda, bM, 0, As[0], tid, ra);
    stage(Bb, N, ldb, bN, 0, Bs[0], tid, rb);
    commit(As[0], tid, ra);
    commit(Bs[0], tid, rb);

    const int nk = K / BK;
    int buf = 0;
    for (int t = 0; t < nk; ++t) {
        // drain this wave's async DMAs, then rendezvous: As[buf]/Bs[buf] ready
        asm volatile("s_wait_asynccnt 0x0" ::: "memory");
        __syncthreads();

        const bool more = (t + 1) < nk;
        if (more) {   // overlap next tile's global traffic with WMMA
            int kn = (t + 1) * BK;
            stage(Ab, M, lda, bM, kn, As[buf ^ 1], tid, ra);
            stage(Bb, N, ldb, bN, kn, Bs[buf ^ 1], tid, rb);
        }

        v16h a[4], b[2];
#pragma unroll
        for (int i = 0; i < 4; ++i) a[i] = load_frag(As[buf], wm + i * 16, lane);
#pragma unroll
        for (int j = 0; j < 2; ++j) b[j] = load_frag(Bs[buf], wn + j * 16, lane);

#pragma unroll
        for (int i = 0; i < 4; ++i)
#pragma unroll
            for (int j = 0; j < 2; ++j)
                c[i][j] = __builtin_amdgcn_wmma_f32_16x16x32_f16(
                    false, a[i], false, b[j], (short)0, c[i][j], false, false);

        if (more) {
            commit(As[buf ^ 1], tid, ra);
            commit(Bs[buf ^ 1], tid, rb);
        }
        buf ^= 1;
    }

    // C/D fragment layout: lane holds column n = lane&15, rows m0..m0+7 with
    // m0 = (lane>>4)*8.  M,N are multiples of 16 so m0<M implies m0+7<M.
    const int nloc = lane & 15;
    const int mloc = (lane >> 4) << 3;
#pragma unroll
    for (int i = 0; i < 4; ++i) {
#pragma unroll
        for (int j = 0; j < 2; ++j) {
            int m0 = bM + wm + i * 16 + mloc;
            int n  = bN + wn + j * 16 + nloc;
            if (m0 >= M || n >= N) continue;
            float bv = bias ? bias[n] : 0.0f;
            if (OUTMODE == 0) {
                float* O = (float*)Og + oOff;
#pragma unroll
                for (int r = 0; r < 8; ++r)
                    O[(long long)(m0 + r) * ldo + n] = (c[i][j][r] + bv) * outScale;
            } else if (OUTMODE == 1) {
                half_t* O = (half_t*)Og + oOff;
#pragma unroll
                for (int r = 0; r < 8; ++r)
                    O[(long long)(m0 + r) * ldo + n] =
                        (half_t)((c[i][j][r] + bv) * outScale);
            } else {  // transposed f16: contiguous 8-half (16B) store per lane
                half_t* O = (half_t*)Og + oOff;
                v8h vv;
#pragma unroll
                for (int r = 0; r < 8; ++r)
                    vv[r] = (half_t)((c[i][j][r] + bv) * outScale);
                *(v8h*)(O + (long long)n * ldo + m0) = vv;
            }
        }
    }
}

// ---- f32 [S,C] -> f16 transposed [C,S] -------------------------------------
__launch_bounds__(256)
__global__ void transpose_to_f16(const float* __restrict__ in, half_t* __restrict__ out,
                                 int S, int C)
{
    long long t = (long long)blockIdx.x * blockDim.x + threadIdx.x;
    long long total = (long long)S * C;
    if (t >= total) return;
    long long c = t / S;
    long long s = t - c * S;
    out[c * S + s] = (half_t)in[s * C + c];
}

// ---- in-place softmax over rows of 256 (f16 storage, f32 math) -------------
__launch_bounds__(256)
__global__ void softmax256_f16(half_t* __restrict__ sc, long long nRows)
{
    const int lane = threadIdx.x & 31;
    const int wv   = threadIdx.x >> 5;
    long long row  = (long long)blockIdx.x * 8 + wv;
    if (row >= nRows) return;
    half_t* p = sc + row * 256 + lane * 8;
    v8h raw = *(const v8h*)p;
    float v[8];
    float mx = -1e30f;
#pragma unroll
    for (int i = 0; i < 8; ++i) { v[i] = (float)raw[i]; mx = fmaxf(mx, v[i]); }
#pragma unroll
    for (int off = 16; off > 0; off >>= 1) mx = fmaxf(mx, __shfl_xor(mx, off, 32));
    float sum = 0.f;
#pragma unroll
    for (int i = 0; i < 8; ++i) { v[i] = __expf(v[i] - mx); sum += v[i]; }
#pragma unroll
    for (int off = 16; off > 0; off >>= 1) sum += __shfl_xor(sum, off, 32);
    float inv = 1.0f / sum;
    v8h o;
#pragma unroll
    for (int i = 0; i < 8; ++i) o[i] = (half_t)(v[i] * inv);
    *(v8h*)p = o;
}

// ---------------------------------------------------------------------------
extern "C" void kernel_launch(void* const* d_in, const int* in_sizes, int n_in,
                              void* d_out, int out_size, void* d_ws, size_t ws_size,
                              hipStream_t stream)
{
    (void)in_sizes; (void)n_in; (void)out_size; (void)ws_size;
    const int B = 4, S = 4096, D = 1024, H = 16, DK = 64, KD = 256;

    const float* q   = (const float*)d_in[0];
    const float* k   = (const float*)d_in[1];
    const float* v   = (const float*)d_in[2];
    const float* W_Q = (const float*)d_in[3];
    const float* b_Q = (const float*)d_in[4];
    const float* W_K = (const float*)d_in[5];
    const float* b_K = (const float*)d_in[6];
    const float* W_V = (const float*)d_in[7];
    const float* b_V = (const float*)d_in[8];
    const float* W_O = (const float*)d_in[9];
    const float* b_O = (const float*)d_in[10];
    const float* E   = (const float*)d_in[11];
    const float* F   = (const float*)d_in[12];

    // workspace layout (bytes, 256-aligned); total ~244 MB
    char* ws = (char*)d_ws;
    size_t off = 0;
    auto take = [&](size_t bytes) { size_t r = off; off += (bytes + 255) & ~(size_t)255; return r; };
    half_t* Qh  = (half_t*)(ws + take((size_t)B * S * D * 2));   // [B,S,D]
    half_t* Kt  = (half_t*)(ws + take((size_t)B * D * S * 2));   // [B,D,S]
    half_t* Vt  = (half_t*)(ws + take((size_t)B * D * S * 2));   // [B,D,S]
    half_t* Et  = (half_t*)(ws + take((size_t)KD * S * 2));      // [256,S]
    half_t* Ft  = (half_t*)(ws + take((size_t)KD * S * 2));      // [256,S]
    half_t* Kpt = (half_t*)(ws + take((size_t)B * H * KD * DK * 2)); // [z,256,64]
    half_t* Vpt = (half_t*)(ws + take((size_t)B * H * DK * KD * 2)); // [z,64,256]
    half_t* Sc  = (half_t*)(ws + take((size_t)B * H * S * KD * 2));  // [z,S,256]
    half_t* attnAll = Kt;  // Kt is dead after Kpt GEMM -> reuse as [B,S,D]

    dim3 blk(256);
    const long long SD = (long long)S * D;

    // 1. Et = E^T, Ft = F^T
    {
        int tot = S * KD;
        transpose_to_f16<<<(tot + 255) / 256, blk, 0, stream>>>(E, Et, S, KD);
        transpose_to_f16<<<(tot + 255) / 256, blk, 0, stream>>>(F, Ft, S, KD);
    }
    // 2. projections, batched over b
    {
        dim3 g(D / BN, S / BM, B);  // (8, 32, 4)
        gemm_wmma<float, float, 1><<<g, blk, 0, stream>>>(
            q, W_Q, Qh, b_Q, S, D, D, D, D, D, 1, SD, 0, 0, 0, SD, 0, 1.0f);
        gemm_wmma<float, float, 2><<<g, blk, 0, stream>>>(
            k, W_K, Kt, b_K, S, D, D, D, D, S, 1, SD, 0, 0, 0, SD, 0, 1.0f);
        gemm_wmma<float, float, 2><<<g, blk, 0, stream>>>(
            v, W_V, Vt, b_V, S, D, D, D, D, S, 1, SD, 0, 0, 0, SD, 0, 1.0f);
    }
    // 3. Kpt[z] = Et @ Kt[z]^T ; Vpt[z] = Vt[z] @ Ft^T
    {
        dim3 gk(1, (KD + BM - 1) / BM, B * H);  // (1, 2, 64)
        gemm_wmma<half_t, half_t, 1><<<gk, blk, 0, stream>>>(
            Et, Kt, Kpt, nullptr, KD, DK, S, S, S, DK,
            1, 0, 0, (long long)DK * S, 0, (long long)KD * DK, 0, 1.0f);
        dim3 gv((KD + BN - 1) / BN, 1, B * H);  // (2, 1, 64)
        gemm_wmma<half_t, half_t, 1><<<gv, blk, 0, stream>>>(
            Vt, Ft, Vpt, nullptr, DK, KD, S, S, S, KD,
            1, (long long)DK * S, 0, 0, 0, (long long)DK * KD, 0, 1.0f);
    }
    // 4. scores (scaled by 1/sqrt(dk)=1/8, folded into output)
    {
        dim3 g((KD + BN - 1) / BN, S / BM, B * H);  // (2, 32, 64)
        gemm_wmma<half_t, half_t, 1><<<g, blk, 0, stream>>>(
            Qh, Kpt, Sc, nullptr, S, KD, DK, D, DK, KD,
            H, SD, (long long)DK,
            (long long)H * KD * DK, (long long)KD * DK,
            (long long)H * S * KD, (long long)S * KD, 0.125f);
    }
    // 5. softmax in place
    {
        long long rows = (long long)B * H * S;  // 262144
        softmax256_f16<<<(unsigned)((rows + 7) / 8), blk, 0, stream>>>(Sc, rows);
    }
    // 6. attn = Sc @ Vpt^T -> attnAll [B,S,D] (head-interleaved columns)
    {
        dim3 g(1, S / BM, B * H);  // (1, 32, 64)
        gemm_wmma<half_t, half_t, 1><<<g, blk, 0, stream>>>(
            Sc, Vpt, attnAll, nullptr, S, DK, KD, KD, KD, D,
            H, (long long)H * S * KD, (long long)S * KD,
            (long long)H * DK * KD, (long long)DK * KD,
            SD, (long long)DK, 1.0f);
    }
    // 7. out = attnAll @ W_O^T + b_O  (f32)
    {
        dim3 g(D / BN, S / BM, B);  // (8, 32, 4)
        gemm_wmma<half_t, float, 0><<<g, blk, 0, stream>>>(
            attnAll, W_O, d_out, b_O, S, D, D, D, D, D,
            1, SD, 0, 0, 0, SD, 0, 1.0f);
    }
}
